// Model_50706383896915
// MI455X (gfx1250) — matrix operand order repaired and checked by reading, not compiled, not run on hardware
//
#include <hip/hip_runtime.h>
#include <hip/hip_bf16.h>

#define BATCH 512
#define LCH 2713
#define ZSEQ 58
#define GDIM 8
#define FFD 2048
#define NLAY 6
#define RNAIN 1018
#define LIN1D 256
#define LIN2D 64
#define CATD 528
#define C1D 64
#define C2D 8
#define XSZ (ZSEQ * GDIM)   // 464
#define CHUNK 64
#define NCH ((LCH + CHUNK - 1) / CHUNK)   // 43 chunks (last has 25 rows)

typedef __attribute__((ext_vector_type(16))) __bf16 v16bf;
typedef __attribute__((ext_vector_type(8)))  float  v8f;

// ---------------------------------------------------------------------------
// Embedding: x0[b,z,g] = sum_l chem_w[g,l]*drug[b,l,z] + pos_w[g,z]
// Bandwidth-bound (322 MB of drug, HBM floor ~14us) -> double-buffered
// GLOBAL_LOAD_ASYNC_TO_LDS_B64 pipeline (ASYNCcnt), compute overlapped with
// the in-flight copy of the next chunk.
// ---------------------------------------------------------------------------
__global__ __launch_bounds__(256) void embed_kernel(
    const float* __restrict__ drug, const float* __restrict__ chem_w,
    const float* __restrict__ pos_w, float* __restrict__ x0) {
  const int b = blockIdx.x;
  const int tid = threadIdx.x;
  __shared__ float ld[2][CHUNK * ZSEQ];
  __shared__ float lc[GDIM][CHUNK];
  float acc0 = 0.f, acc1 = 0.f;
  const int o0 = tid, o1 = tid + 256;
  const int z0 = o0 >> 3, g0 = o0 & 7;
  const int z1 = o1 >> 3, g1 = o1 & 7;
  const float* dbase = drug + (size_t)b * (size_t)(LCH * ZSEQ);

  // prologue: async-copy chunk 0 (always full) into buffer 0
  {
    const unsigned lbase = (unsigned)(uintptr_t)&ld[0][0];
    for (int i = tid; i < (CHUNK * ZSEQ) >> 1; i += 256) {
      const unsigned long long src =
          (unsigned long long)(uintptr_t)(dbase + 2 * i);
      const unsigned dst = lbase + 8u * i;
      asm volatile("global_load_async_to_lds_b64 %0, %1, off"
                   :: "v"(dst), "v"(src) : "memory");
    }
  }

  for (int ch = 0; ch < NCH; ++ch) {
    const int l0 = ch * CHUNK;
    const int cl = (LCH - l0 < CHUNK) ? (LCH - l0) : CHUNK;
    const int cur = ch & 1;

    // current buffer's copy (issued last iteration) has landed; sync all waves
    asm volatile("s_wait_asynccnt 0" ::: "memory");
    __syncthreads();

    // kick off next chunk's copy into the other buffer (overlaps compute)
    if (ch + 1 < NCH) {
      const int nl0 = l0 + CHUNK;
      const int ncl = (LCH - nl0 < CHUNK) ? (LCH - nl0) : CHUNK;
      const int ncnt = ncl * ZSEQ;                 // always even (Z=58)
      const unsigned lbase = (unsigned)(uintptr_t)&ld[cur ^ 1][0];
      const float* gsrc = dbase + (size_t)nl0 * ZSEQ;  // 8B-aligned
      for (int i = tid; i < (ncnt >> 1); i += 256) {
        const unsigned long long src =
            (unsigned long long)(uintptr_t)(gsrc + 2 * i);
        const unsigned dst = lbase + 8u * i;
        asm volatile("global_load_async_to_lds_b64 %0, %1, off"
                     :: "v"(dst), "v"(src) : "memory");
      }
    }

    // chem_w slice for this chunk (tiny, L2-resident)
    for (int idx = tid; idx < GDIM * CHUNK; idx += 256) {
      const int g = idx >> 6, j = idx & 63;
      lc[g][j] = (j < cl) ? chem_w[g * LCH + l0 + j] : 0.f;
    }
    __syncthreads();

    #pragma unroll 4
    for (int j = 0; j < cl; ++j) {
      acc0 = fmaf(lc[g0][j], ld[cur][j * ZSEQ + z0], acc0);
      if (o1 < XSZ) acc1 = fmaf(lc[g1][j], ld[cur][j * ZSEQ + z1], acc1);
    }
  }
  x0[b * XSZ + o0] = acc0 + pos_w[g0 * ZSEQ + z0];
  if (o1 < XSZ) x0[b * XSZ + o1] = acc1 + pos_w[g1 * ZSEQ + z1];
}

// ---------------------------------------------------------------------------
// RNA branch GEMM1: r1[512,256] = rna @ lin1_w^T + b  (K=1018)
// ---------------------------------------------------------------------------
__global__ __launch_bounds__(256) void rna1_kernel(
    const float* __restrict__ rna, const float* __restrict__ w,
    const float* __restrict__ bias, float* __restrict__ r1) {
  const int bb0 = blockIdx.x * 8;
  const int t = threadIdx.x;
  __shared__ float xr[8][RNAIN];
  for (int idx = t; idx < 8 * RNAIN; idx += 256)
    ((float*)xr)[idx] = rna[(size_t)bb0 * RNAIN + idx];  // 8 contiguous rows
  __syncthreads();
  float acc[8];
  #pragma unroll
  for (int r = 0; r < 8; ++r) acc[r] = bias[t];
  for (int k = 0; k < RNAIN; ++k) {
    const float wv = w[t * RNAIN + k];
    #pragma unroll
    for (int r = 0; r < 8; ++r) acc[r] = fmaf(wv, xr[r][k], acc[r]);
  }
  #pragma unroll
  for (int r = 0; r < 8; ++r) r1[(bb0 + r) * LIN1D + t] = acc[r];
}

// RNA branch GEMM2: r2[512,64] = r1 @ lin2_w^T + b  (K=256)
__global__ __launch_bounds__(64) void rna2_kernel(
    const float* __restrict__ r1, const float* __restrict__ w,
    const float* __restrict__ bias, float* __restrict__ r2) {
  const int b = blockIdx.x, t = threadIdx.x;
  __shared__ float row[LIN1D];
  for (int idx = t; idx < LIN1D; idx += 64) row[idx] = r1[b * LIN1D + idx];
  __syncthreads();
  float acc = bias[t];
  for (int k = 0; k < LIN1D; ++k) acc = fmaf(w[t * LIN1D + k], row[k], acc);
  r2[b * LIN2D + t] = acc;
}

// ---------------------------------------------------------------------------
// One fused transformer layer per (b, workgroup): qkv -> attn(HD=1) -> proj
// -> LN1 -> FFN via bf16 WMMA -> LN2.  128 threads = 4 waves = 4 M-tiles.
// ---------------------------------------------------------------------------
__global__ __launch_bounds__(128) void layer_kernel(
    const float* __restrict__ x_in, float* __restrict__ x_out,
    const float* __restrict__ qkv_w, const float* __restrict__ qkv_b,
    const float* __restrict__ out_w, const float* __restrict__ out_b,
    const float* __restrict__ ln1_g, const float* __restrict__ ln1_b,
    const float* __restrict__ ff1_w, const float* __restrict__ ff1_b,
    const float* __restrict__ ff2_w, const float* __restrict__ ff2_b,
    const float* __restrict__ ln2_g, const float* __restrict__ ln2_b,
    int layer) {
  const int b = blockIdx.x;
  const int tid = threadIdx.x;

  qkv_w += layer * 3 * GDIM * GDIM;  qkv_b += layer * 3 * GDIM;
  out_w += layer * GDIM * GDIM;      out_b += layer * GDIM;
  ln1_g += layer * GDIM;             ln1_b += layer * GDIM;
  ff1_w += (size_t)layer * FFD * GDIM;  ff1_b += layer * FFD;
  ff2_w += (size_t)layer * GDIM * FFD;  ff2_b += layer * GDIM;
  ln2_g += layer * GDIM;             ln2_b += layer * GDIM;

  __shared__ float  xbuf[ZSEQ][GDIM];
  __shared__ float  qk[ZSEQ][3 * GDIM];
  __shared__ float  att[ZSEQ][GDIM];
  __shared__ float  obuf[64][GDIM];
  __shared__ __bf16 hbuf[4][16][34];   // per-wave 16x32 hidden tile (+pad)

  // ---- load x[b] ----
  for (int idx = tid; idx < XSZ; idx += 128)
    ((float*)xbuf)[idx] = x_in[b * XSZ + idx];
  __syncthreads();

  // ---- qkv projection (K=8, VALU) ----
  for (int idx = tid; idx < ZSEQ * 3 * GDIM; idx += 128) {
    const int z = idx / 24, j = idx % 24;
    float s = qkv_b[j];
    #pragma unroll
    for (int g = 0; g < GDIM; ++g) s = fmaf(qkv_w[j * GDIM + g], xbuf[z][g], s);
    qk[z][j] = s;
  }
  __syncthreads();

  // ---- attention, head_dim = 1 (scale = 1): scalar softmax over t ----
  for (int idx = tid; idx < XSZ; idx += 128) {
    const int z = idx >> 3, h = idx & 7;
    const float q = qk[z][h];
    float m = -1e30f;
    for (int t = 0; t < ZSEQ; ++t) m = fmaxf(m, q * qk[t][GDIM + h]);
    float se = 0.f, sv = 0.f;
    for (int t = 0; t < ZSEQ; ++t) {
      const float e = __expf(q * qk[t][GDIM + h] - m);
      se += e;
      sv = fmaf(e, qk[t][2 * GDIM + h], sv);
    }
    att[z][h] = sv / se;
  }
  __syncthreads();

  // ---- out-proj + residual + LN1 (row per thread) ----
  if (tid < ZSEQ) {
    const int z = tid;
    float y[GDIM], mu = 0.f;
    #pragma unroll
    for (int g = 0; g < GDIM; ++g) {
      float o = out_b[g];
      #pragma unroll
      for (int gp = 0; gp < GDIM; ++gp)
        o = fmaf(out_w[g * GDIM + gp], att[z][gp], o);
      y[g] = xbuf[z][g] + o;
      mu += y[g];
    }
    mu *= 0.125f;
    float var = 0.f;
    #pragma unroll
    for (int g = 0; g < GDIM; ++g) { const float d = y[g] - mu; var = fmaf(d, d, var); }
    var *= 0.125f;
    const float rs = rsqrtf(var + 1e-5f);
    #pragma unroll
    for (int g = 0; g < GDIM; ++g)
      xbuf[z][g] = (y[g] - mu) * rs * ln1_g[g] + ln1_b[g];
  }
  __syncthreads();

  // ---- FFN: [64x8] @ [8x2048] relu @ [2048x8] via v_wmma_f32_16x16x32_bf16 --
  const int wave  = tid >> 5;
  const int lane  = tid & 31;
  const int lhalf = lane >> 4;     // 0: lanes 0-15, 1: lanes 16-31
  const int lm    = lane & 15;
  const int m0    = wave * 16;

  // A (x rows, 16x32, real K=8): lanes<16 hold K=0..7 in halves 0..7
  v16bf a;
  #pragma unroll
  for (int j = 0; j < 16; ++j) a[j] = (__bf16)0.f;
  {
    const int row = m0 + lm;
    if (lhalf == 0 && row < ZSEQ) {
      #pragma unroll
      for (int j = 0; j < GDIM; ++j) a[j] = (__bf16)xbuf[row][j];
    }
  }

  v8f outc;
  #pragma unroll
  for (int q = 0; q < 8; ++q) outc[q] = 0.f;

  #pragma unroll 1
  for (int c = 0; c < FFD / 32; ++c) {
    // GEMM1: two 16-wide N tiles of the 32-col hidden chunk
    #pragma unroll
    for (int nt = 0; nt < 2; ++nt) {
      const int fbase = c * 32 + nt * 16;
      v16bf bw;   // B 32x16: k=g rows (8 real), n=f cols
      #pragma unroll
      for (int j = 0; j < 16; ++j) bw[j] = (__bf16)0.f;
      if (lhalf == 0) {
        #pragma unroll
        for (int j = 0; j < GDIM; ++j)
          bw[j] = (__bf16)ff1_w[(fbase + lm) * GDIM + j];
      }
      v8f hc;
      #pragma unroll
      for (int q = 0; q < 8; ++q) hc[q] = 0.f;
      hc = __builtin_amdgcn_wmma_f32_16x16x32_bf16(
               false, a, false, bw, (short)0, hc, false, false);
      const float hb = ff1_b[fbase + lm];
      #pragma unroll
      for (int q = 0; q < 8; ++q) {
        const float hv = fmaxf(hc[q] + hb, 0.f);          // bias + relu
        hbuf[wave][q + 8 * lhalf][nt * 16 + lm] = (__bf16)hv;  // D->LDS transpose
      }
    }
    // same-wave DS ops are in-order; fence compiler + wait before re-reading
    asm volatile("s_wait_dscnt 0" ::: "memory");

    // GEMM2: A2 = hidden tile 16x32 (A layout), B2 = ff2^T chunk 32x16
    v16bf a2;
    #pragma unroll
    for (int j = 0; j < 16; ++j) {
      const int k = 8 * lhalf + ((j < 8) ? j : (j + 8));
      a2[j] = hbuf[wave][lm][k];
    }
    v16bf b2;
    #pragma unroll
    for (int j = 0; j < 16; ++j) {
      const float wv = (lm < GDIM) ? ff2_w[lm * FFD + c * 32 + 8 * lhalf + ((j < 8) ? j : (j + 8))] : 0.f;
      b2[j] = (__bf16)wv;
    }
    outc = __builtin_amdgcn_wmma_f32_16x16x32_bf16(
               false, a2, false, b2, (short)0, outc, false, false);
  }

  // D layout -> obuf rows
  #pragma unroll
  for (int q = 0; q < 8; ++q) {
    const int m = q + 8 * lhalf;
    if (lm < GDIM) obuf[m0 + m][lm] = outc[q];
  }
  __syncthreads();

  // ---- bias + residual + LN2 -> x_out ----
  if (tid < ZSEQ) {
    const int z = tid;
    float y[GDIM], mu = 0.f;
    #pragma unroll
    for (int g = 0; g < GDIM; ++g) {
      y[g] = xbuf[z][g] + obuf[z][g] + ff2_b[g];
      mu += y[g];
    }
    mu *= 0.125f;
    float var = 0.f;
    #pragma unroll
    for (int g = 0; g < GDIM; ++g) { const float d = y[g] - mu; var = fmaf(d, d, var); }
    var *= 0.125f;
    const float rs = rsqrtf(var + 1e-5f);
    #pragma unroll
    for (int g = 0; g < GDIM; ++g)
      x_out[b * XSZ + z * GDIM + g] = (y[g] - mu) * rs * ln2_g[g] + ln2_b[g];
  }
}

// ---------------------------------------------------------------------------
// Head: concat(r2, x_flat) -> 64 -> 8 -> 1  (no activations, per reference)
// ---------------------------------------------------------------------------
__global__ __launch_bounds__(64) void head_kernel(
    const float* __restrict__ r2, const float* __restrict__ x,
    const float* __restrict__ c1_w, const float* __restrict__ c1_b,
    const float* __restrict__ c2_w, const float* __restrict__ c2_b,
    const float* __restrict__ c3_w, const float* __restrict__ c3_b,
    float* __restrict__ out) {
  const int b = blockIdx.x, t = threadIdx.x;
  __shared__ float h[CATD];
  __shared__ float h1[C1D];
  __shared__ float h2[C2D];
  for (int idx = t; idx < CATD; idx += 64)
    h[idx] = (idx < LIN2D) ? r2[b * LIN2D + idx] : x[b * XSZ + idx - LIN2D];
  __syncthreads();
  float acc = c1_b[t];
  for (int k = 0; k < CATD; ++k) acc = fmaf(c1_w[t * CATD + k], h[k], acc);
  h1[t] = acc;
  __syncthreads();
  if (t < C2D) {
    float a2 = c2_b[t];
    for (int k = 0; k < C1D; ++k) a2 = fmaf(c2_w[t * C1D + k], h1[k], a2);
    h2[t] = a2;
  }
  __syncthreads();
  if (t == 0) {
    float a3 = c3_b[0];
    #pragma unroll
    for (int k = 0; k < C2D; ++k) a3 = fmaf(c3_w[k], h2[k], a3);
    out[b] = a3;
  }
}

// ---------------------------------------------------------------------------
extern "C" void kernel_launch(void* const* d_in, const int* in_sizes, int n_in,
                              void* d_out, int out_size, void* d_ws, size_t ws_size,
                              hipStream_t stream) {
  (void)in_sizes; (void)n_in; (void)out_size; (void)ws_size;
  const float* rna    = (const float*)d_in[0];
  const float* drug   = (const float*)d_in[1];
  const float* lin1_w = (const float*)d_in[2];
  const float* lin1_b = (const float*)d_in[3];
  const float* lin2_w = (const float*)d_in[4];
  const float* lin2_b = (const float*)d_in[5];
  const float* chem_w = (const float*)d_in[6];
  const float* pos_w  = (const float*)d_in[7];
  const float* qkv_w  = (const float*)d_in[8];
  const float* qkv_b  = (const float*)d_in[9];
  const float* out_w  = (const float*)d_in[10];
  const float* out_b  = (const float*)d_in[11];
  const float* ln1_g  = (const float*)d_in[12];
  const float* ln1_b  = (const float*)d_in[13];
  const float* ff1_w  = (const float*)d_in[14];
  const float* ff1_b  = (const float*)d_in[15];
  const float* ff2_w  = (const float*)d_in[16];
  const float* ff2_b  = (const float*)d_in[17];
  const float* ln2_g  = (const float*)d_in[18];
  const float* ln2_b  = (const float*)d_in[19];
  const float* c1_w   = (const float*)d_in[20];
  const float* c1_b   = (const float*)d_in[21];
  const float* c2_w   = (const float*)d_in[22];
  const float* c2_b   = (const float*)d_in[23];
  const float* c3_w   = (const float*)d_in[24];
  const float* c3_b   = (const float*)d_in[25];
  float* out = (float*)d_out;

  float* ws = (float*)d_ws;
  float* X0 = ws;                       // 512*464
  float* X1 = X0 + BATCH * XSZ;         // 512*464
  float* R1 = X1 + BATCH * XSZ;         // 512*256
  float* R2 = R1 + BATCH * LIN1D;       // 512*64   (total ~2.6 MB)

  embed_kernel<<<BATCH, 256, 0, stream>>>(drug, chem_w, pos_w, X0);
  rna1_kernel<<<BATCH / 8, 256, 0, stream>>>(rna, lin1_w, lin1_b, R1);
  rna2_kernel<<<BATCH, 64, 0, stream>>>(R1, lin2_w, lin2_b, R2);

  float* xin = X0;
  float* xout = X1;
  for (int i = 0; i < NLAY; ++i) {
    layer_kernel<<<BATCH, 128, 0, stream>>>(xin, xout, qkv_w, qkv_b,
        out_w, out_b, ln1_g, ln1_b, ff1_w, ff1_b, ff2_w, ff2_b,
        ln2_g, ln2_b, i);
    float* tmp = xin; xin = xout; xout = tmp;
  }
  head_kernel<<<BATCH, 64, 0, stream>>>(R2, xin, c1_w, c1_b, c2_w, c2_b,
                                        c3_w, c3_b, out);
}
